// DVAE_hybirdLoss_5686536699907
// MI455X (gfx1250) — matrix operand order repaired and checked by reading, not compiled, not run on hardware
//
#include <hip/hip_runtime.h>
#include <hip/hip_bf16.h>

// ---------------- problem constants ----------------
#define BATCH 4096
#define MAXN  24
#define NVT   26
#define HS    501
#define HSP   512      // HS padded to multiple of 32 for WMMA K / 16 for N
#define NZ    56

typedef __attribute__((ext_vector_type(16))) __bf16 v16bf;
typedef __attribute__((ext_vector_type(8)))  float  v8f;

union Frag { v16bf v; uint4 u[2]; };

// ---------------- bf16 helpers (bit-exact RNE, no __bf16 arithmetic) -------
static __device__ inline __bf16 f2bf(float f) {
    unsigned u = __builtin_bit_cast(unsigned, f);
    unsigned lsb = (u >> 16) & 1u;
    u += 0x7fffu + lsb;                      // round to nearest even
    unsigned short h = (unsigned short)(u >> 16);
    return __builtin_bit_cast(__bf16, h);
}
static __device__ inline float bf2f(__bf16 b) {
    unsigned short h = __builtin_bit_cast(unsigned short, b);
    unsigned u = ((unsigned)h) << 16;
    return __builtin_bit_cast(float, u);
}
static __device__ inline float sigmf(float x) { return 1.f / (1.f + __expf(-x)); }
static __device__ inline float tanh_fast(float x) {
    float e = __expf(2.f * x);
    return (e - 1.f) / (e + 1.f);
}

// ---------------- WMMA fragment loaders (CDNA5 16-bit layouts) -------------
// A 16x32 (MxK) from LDS tile (row-major, stride HSP):
//   lane<16  -> row=lane,    K = k0 + {0..7, 16..23}
//   lane>=16 -> row=lane-16, K = k0 + {8..15, 24..31}
static __device__ inline v16bf load_a_lds(const __bf16* tile, int lane, int k0) {
    int row = lane & 15;
    int kb  = (lane < 16) ? 0 : 8;
    const __bf16* p = tile + row * HSP + k0 + kb;
    Frag f;
    f.u[0] = *reinterpret_cast<const uint4*>(p);       // K = k0+kb .. +7
    f.u[1] = *reinterpret_cast<const uint4*>(p + 16);  // K = k0+kb+16 .. +23
    return f.v;
}
// B 32x16 (KxN), weights stored column-major: WT[n*HSP + k]
// lane<16 -> col=n0+lane, K = k0+{0..7,16..23} ; lane>=16 -> col=n0+lane-16, K = k0+{8..15,24..31}
static __device__ inline v16bf load_b(const __bf16* wt, int lane, int n0, int k0) {
    int col = n0 + (lane & 15);
    int kb  = (lane < 16) ? 0 : 8;
    const __bf16* p = wt + (size_t)col * HSP + (size_t)(k0 + kb);
    Frag f;
    f.u[0] = *reinterpret_cast<const uint4*>(p);
    f.u[1] = *reinterpret_cast<const uint4*>(p + 16);
    return f.v;
}
static __device__ inline v8f wmma_bf16(v16bf a, v16bf b, v8f c) {
    return __builtin_amdgcn_wmma_f32_16x16x32_bf16(false, a, false, b,
                                                   (short)0, c, false, false);
}

// Cooperative copy of a 16 x HSP bf16 A-tile (16 KB) global -> LDS.
// 256 threads, 4 x 16-byte chunks each; 64 chunks (1024 B) per row.
static __device__ inline void stage_a_tile(__bf16* tile, const __bf16* gbase,
                                           size_t row_stride) {
    for (int i = 0; i < 4; ++i) {
        int c   = threadIdx.x + i * 256;       // 0..1023
        int row = c >> 6;                      // /64
        int col = (c & 63) * 8;                // element offset
        const uint4* src = reinterpret_cast<const uint4*>(
            gbase + (size_t)row * row_stride + col);
        *reinterpret_cast<uint4*>(tile + row * HSP + col) = *src;
    }
    __syncthreads();
}

// ---------------- kernel 1: weight conversion + zero init ------------------
// WT layout: [WgT | WmT | WrT | WzT | WnT], each HSP*HSP bf16, col-major (n*HSP+k)
__global__ __launch_bounds__(256) void prep_kernel(
    const float* __restrict__ Wg, const float* __restrict__ Wm,
    const float* __restrict__ W_hh,
    __bf16* __restrict__ WT, float* __restrict__ HaggF,
    __bf16* __restrict__ HaggBF) {
    const int NW = 5 * HSP * HSP;
    int idx = blockIdx.x * 256 + threadIdx.x;
    if (idx < NW) {
        int chunk = idx / (HSP * HSP);
        int r     = idx % (HSP * HSP);
        int n = r / HSP;   // output column
        int k = r % HSP;   // K index
        float f = 0.f;
        if (k < HS && n < HS) {
            if (chunk == 0)      f = Wg[k * HS + n];                 // Wg_h[k,n]
            else if (chunk == 1) f = Wm[k * HS + n];                 // Wm_h[k,n]
            else {
                int c = chunk - 2;                                   // r,z,n gates
                f = W_hh[(size_t)(c * HS + n) * HS + k];             // W_hh.T[k, c*HS+n]
            }
        }
        WT[idx] = f2bf(f);
    } else {
        int i = idx - NW;
        if (i < BATCH * HSP) { HaggF[i] = 0.f; HaggBF[i] = f2bf(0.f); }
    }
}

// ---------------- kernel 2: per-step elementwise aggregation ---------------
// Hagg[b,h] = sum_{u<v, adj[b,u,v]} sig(w*G[b,u,h]+Wg_id[u,h]+bg[h]) * (w*M[b,u,h]+Wm_id[u,h])
__global__ __launch_bounds__(256) void agg_kernel(
    const int* __restrict__ adj, const int* __restrict__ wts,
    const __bf16* __restrict__ Gbf, const __bf16* __restrict__ Mbf,
    const float* __restrict__ Wg, const float* __restrict__ Wm,
    const float* __restrict__ bg,
    float* __restrict__ HaggF, __bf16* __restrict__ HaggBF, int v) {
    int idx = blockIdx.x * 256 + threadIdx.x;
    int b = idx >> 9;          // / HSP
    int h = idx & (HSP - 1);   // % HSP
    float s = 0.f;
    if (h < HS) {
        float bgv = bg[h];
        for (int u = 0; u < v; ++u) {
            int a = adj[(b * MAXN + u) * MAXN + v];
            if (a) {
                float w = (float)wts[(b * MAXN + u) * MAXN + v] * 10.f + 1.f;
                size_t gi = ((size_t)b * MAXN + u) * HSP + h;
                float gate = sigmf(bf2f(Gbf[gi]) * w + Wg[(size_t)(HS + u) * HS + h] + bgv);
                float mapd = bf2f(Mbf[gi]) * w + Wm[(size_t)(HS + u) * HS + h];
                s += gate * mapd;
            }
        }
    }
    HaggF[idx]  = s;
    HaggBF[idx] = f2bf(s);
}

// ---------------- kernel 3: GRU cell (WMMA: Hagg @ W_hh.T) -----------------
__global__ __launch_bounds__(256) void gru_kernel(
    const __bf16* __restrict__ HaggBF, const float* __restrict__ HaggF,
    const __bf16* __restrict__ WT,      // +2..4 chunks are Wr/Wz/Wn
    const int* __restrict__ types, const float* __restrict__ W_ih,
    const float* __restrict__ b_ih, const float* __restrict__ b_hh,
    __bf16* __restrict__ Hbf, float* __restrict__ HvF, int v) {
    __shared__ __align__(16) __bf16 Atile[16 * HSP];   // 16 KB
    int lane = threadIdx.x & 31;
    int wave = threadIdx.x >> 5;
    int b0 = blockIdx.x * 16;
    int n0 = (blockIdx.y * 8 + wave) * 16;
    const __bf16* WrT = WT + (size_t)2 * HSP * HSP;
    const __bf16* WzT = WT + (size_t)3 * HSP * HSP;
    const __bf16* WnT = WT + (size_t)4 * HSP * HSP;

    stage_a_tile(Atile, HaggBF + (size_t)b0 * HSP, HSP);

    v8f ar = {}, az = {}, an = {};
    for (int k0 = 0; k0 < HSP; k0 += 32) {
        v16bf a  = load_a_lds(Atile, lane, k0);
        v16bf br = load_b(WrT, lane, n0, k0);
        v16bf bz = load_b(WzT, lane, n0, k0);
        v16bf bn = load_b(WnT, lane, n0, k0);
        ar = wmma_bf16(a, br, ar);
        az = wmma_bf16(a, bz, az);
        an = wmma_bf16(a, bn, an);
    }
    int h  = n0 + (lane & 15);
    int mb = (lane < 16) ? 0 : 8;
    for (int r = 0; r < 8; ++r) {
        int b = b0 + r + mb;
        float hv = 0.f;
        if (h < HS) {
            int t = types[b * MAXN + v];
            float ir  = W_ih[(size_t)h * NVT + t]            + b_ih[h];
            float iz  = W_ih[(size_t)(HS + h) * NVT + t]     + b_ih[HS + h];
            float inn = W_ih[(size_t)(2 * HS + h) * NVT + t] + b_ih[2 * HS + h];
            float hr = ar[r] + b_hh[h];
            float hz = az[r] + b_hh[HS + h];
            float hn = an[r] + b_hh[2 * HS + h];
            float rg = sigmf(ir + hr);
            float zg = sigmf(iz + hz);
            float nn = tanh_fast(inn + rg * hn);
            float hprev = HaggF[(size_t)b * HSP + h];
            hv = (1.f - zg) * nn + zg * hprev;
        }
        Hbf[((size_t)b * MAXN + v) * HSP + h] = f2bf(hv);
        HvF[(size_t)b * HSP + h] = hv;
    }
}

// ---------------- kernel 4: G/M precompute (WMMA: Hv @ Wg_h, Hv @ Wm_h) ----
__global__ __launch_bounds__(256) void gm_kernel(
    const __bf16* __restrict__ Hbf, const __bf16* __restrict__ WT,
    __bf16* __restrict__ Gbf, __bf16* __restrict__ Mbf, int v) {
    __shared__ __align__(16) __bf16 Atile[16 * HSP];   // 16 KB
    int lane = threadIdx.x & 31;
    int wave = threadIdx.x >> 5;
    int b0 = blockIdx.x * 16;
    int n0 = (blockIdx.y * 8 + wave) * 16;
    const __bf16* WgT = WT;
    const __bf16* WmT = WT + (size_t)HSP * HSP;

    stage_a_tile(Atile, Hbf + ((size_t)b0 * MAXN + v) * HSP,
                 (size_t)MAXN * HSP);

    v8f g = {}, m = {};
    for (int k0 = 0; k0 < HSP; k0 += 32) {
        v16bf a  = load_a_lds(Atile, lane, k0);
        v16bf bg = load_b(WgT, lane, n0, k0);
        v16bf bm = load_b(WmT, lane, n0, k0);
        g = wmma_bf16(a, bg, g);
        m = wmma_bf16(a, bm, m);
    }
    int h  = n0 + (lane & 15);
    int mb = (lane < 16) ? 0 : 8;
    for (int r = 0; r < 8; ++r) {
        int b = b0 + r + mb;
        size_t idx = ((size_t)b * MAXN + v) * HSP + h;
        Gbf[idx] = f2bf(g[r]);
        Mbf[idx] = f2bf(m[r]);
    }
}

// ---------------- kernel 5: final projections ------------------------------
__global__ __launch_bounds__(256) void final_kernel(
    const float* __restrict__ HvF,
    const float* __restrict__ W1, const float* __restrict__ b1,
    const float* __restrict__ W2, const float* __restrict__ b2,
    float* __restrict__ out) {
    int idx = blockIdx.x * 256 + threadIdx.x;
    if (idx >= BATCH * NZ) return;
    int b = idx / NZ;
    int j = idx % NZ;
    float s1 = b1[j], s2 = b2[j];
    const float* hrow = HvF + (size_t)b * HSP;
    for (int h = 0; h < HS; ++h) {
        float x = hrow[h];
        s1 = fmaf(x, W1[(size_t)h * NZ + j], s1);
        s2 = fmaf(x, W2[(size_t)h * NZ + j], s2);
    }
    out[idx] = s1;                      // mu
    out[BATCH * NZ + idx] = s2;         // logvar
}

// ---------------- host launcher --------------------------------------------
extern "C" void kernel_launch(void* const* d_in, const int* in_sizes, int n_in,
                              void* d_out, int out_size, void* d_ws, size_t ws_size,
                              hipStream_t stream) {
    (void)in_sizes; (void)n_in; (void)out_size; (void)ws_size;
    const int*   types = (const int*)d_in[0];
    const int*   adj   = (const int*)d_in[1];
    const int*   wts   = (const int*)d_in[2];
    const float* W_ih  = (const float*)d_in[3];
    const float* W_hh  = (const float*)d_in[4];
    const float* b_ih  = (const float*)d_in[5];
    const float* b_hh  = (const float*)d_in[6];
    const float* Wg    = (const float*)d_in[7];
    const float* bg    = (const float*)d_in[8];
    const float* Wm    = (const float*)d_in[9];
    const float* W1    = (const float*)d_in[10];
    const float* b1    = (const float*)d_in[11];
    const float* W2    = (const float*)d_in[12];
    const float* b2    = (const float*)d_in[13];
    float* out = (float*)d_out;

    char* p = (char*)d_ws;
    __bf16* Hbf    = (__bf16*)p; p += (size_t)BATCH * MAXN * HSP * 2;
    __bf16* Gbf    = (__bf16*)p; p += (size_t)BATCH * MAXN * HSP * 2;
    __bf16* Mbf    = (__bf16*)p; p += (size_t)BATCH * MAXN * HSP * 2;
    __bf16* WT     = (__bf16*)p; p += (size_t)5 * HSP * HSP * 2;
    float*  HaggF  = (float*)p;  p += (size_t)BATCH * HSP * 4;
    __bf16* HaggBF = (__bf16*)p; p += (size_t)BATCH * HSP * 2;
    float*  HvF    = (float*)p;  p += (size_t)BATCH * HSP * 4;

    // 1) weight conversion (+ zero Hagg for v=0)
    {
        int total = 5 * HSP * HSP + BATCH * HSP;
        prep_kernel<<<(total + 255) / 256, 256, 0, stream>>>(Wg, Wm, W_hh, WT,
                                                             HaggF, HaggBF);
    }

    dim3 gemm_grid(BATCH / 16, HSP / 128);   // 256 x 4, 8 waves/WG, 16 cols/wave
    int agg_blocks = (BATCH * HSP) / 256;    // 8192

    // 2) sequential node loop
    for (int v = 0; v < MAXN; ++v) {
        if (v > 0)
            agg_kernel<<<agg_blocks, 256, 0, stream>>>(adj, wts, Gbf, Mbf, Wg,
                                                       Wm, bg, HaggF, HaggBF, v);
        gru_kernel<<<gemm_grid, 256, 0, stream>>>(HaggBF, HaggF, WT, types,
                                                  W_ih, b_ih, b_hh, Hbf, HvF, v);
        if (v < MAXN - 1)
            gm_kernel<<<gemm_grid, 256, 0, stream>>>(Hbf, WT, Gbf, Mbf, v);
    }

    // 3) mu / logvar projections
    final_kernel<<<(BATCH * NZ + 255) / 256, 256, 0, stream>>>(HvF, W1, b1,
                                                               W2, b2, out);
}
